// MambaBlock_1726576854837
// MI455X (gfx1250) — matrix operand order repaired and checked by reading, not compiled, not run on hardware
//
#include <hip/hip_runtime.h>
#include <hip/hip_bf16.h>
#include <math.h>

// Problem constants (from reference)
#define B_  2
#define L_  1024
#define DM_ 1024
#define DI_ 2048
#define DS_ 16
#define DR_ 64
#define BL_ (B_ * L_)
// Scan chunking
#define NC_ 32
#define CL_ (L_ / NC_)   // 32 steps per chunk

typedef __attribute__((ext_vector_type(16))) __bf16   v16bf;
typedef __attribute__((ext_vector_type(8)))  float    v8f;
typedef __attribute__((ext_vector_type(4)))  unsigned u32x4;
typedef __attribute__((ext_vector_type(8)))  int      i32x8;
typedef __attribute__((ext_vector_type(4)))  int      i32x4;

// ---------------------------------------------------------------------------
// fp32 -> bf16 conversion
// ---------------------------------------------------------------------------
__global__ void cvt_f32_to_bf16(const float* __restrict__ in,
                                __bf16* __restrict__ out, int n) {
  int i = blockIdx.x * blockDim.x + threadIdx.x;
  if (i < n) out[i] = (__bf16)in[i];
}

// ---------------------------------------------------------------------------
// TDM: DMA one 128-row x 32-col bf16 tile (row stride K elems) into LDS.
// LDS is padded: every 16 DWORDs (one 32-elem row = 64B) insert 8 DWORDs (32B)
// -> LDS row stride 48 bf16 elements. OOB rows (>= rows) read as zero.
// D# layout per CDNA5 ISA 8.3/8.4 (group0 128b, group1 256b).
// 6-arg builtin: (u32x4 g0, i32x8 g1, i32x4 g2, i32x4 g3, i32x8 extra, i32 cpol)
// ---------------------------------------------------------------------------
#define LDS_STRIDE 48
#define TILE_ELEMS (128 * LDS_STRIDE)

__device__ __forceinline__ void tdm_load_tile(unsigned lds_off, const void* gp,
                                              unsigned rows, unsigned k_stride) {
  unsigned long long ga = (unsigned long long)(size_t)gp;
  u32x4 g0;
  g0[0] = 1u;                                       // count=1, user descriptor
  g0[1] = lds_off;                                  // lds_addr [63:32]
  g0[2] = (unsigned)ga;                             // global_addr [95:64]
  g0[3] = (unsigned)(ga >> 32) | (2u << 30);        // global_addr hi + type=2
  i32x8 g1;
  g1[0] = (int)((1u << 16)    // data_size = 1 -> 2 bytes
              | (1u << 20)    // pad_enable
              | (3u << 22)    // pad_interval = 16 DWORDs (one 64B row)
              | (7u << 25));  // pad_amount   = 8 DWORDs (32B)
  g1[1] = (int)(32u << 16);                 // tensor_dim0[15:0] = 32 (in [79:48])
  g1[2] = (int)((rows & 0xffffu) << 16);    // tensor_dim1 lo (in [111:80])
  g1[3] = (int)(32u << 16);                 // tile_dim0 = 32 (in [127:112])
  g1[4] = 128;                              // tile_dim1 = 128 rows
  g1[5] = (int)k_stride;                    // tensor_dim0_stride = K elems
  g1[6] = 0;
  g1[7] = 0;
  i32x4 gz4 = {0, 0, 0, 0};
  i32x8 gz8 = {0, 0, 0, 0, 0, 0, 0, 0};
  __builtin_amdgcn_tensor_load_to_lds(g0, g1, gz4, gz4, gz8, 0);
}

// ---------------------------------------------------------------------------
// Generic WMMA BF16 GEMM:  C[M,N] = A[M,K] * B[N,K]^T   (fp32 accumulate)
// 128x128 tile / workgroup (256 thr = 8 waves), K step 32, TDM double-buffer.
// Wave (wm in 0..3, wn in 0..1) owns a 32x64 sub-tile = 2x4 WMMA tiles.
// ---------------------------------------------------------------------------
__global__ __launch_bounds__(256)
void wmma_gemm_bf16(const __bf16* __restrict__ A, const __bf16* __restrict__ Bw,
                    float* __restrict__ C, int M, int N, int K) {
  __shared__ __bf16 lsA[2][TILE_ELEMS];
  __shared__ __bf16 lsB[2][TILE_ELEMS];

  const int tid  = threadIdx.x;
  const int lane = tid & 31;
  const int wave = tid >> 5;
  const int wm   = wave & 3;   // 4 waves along M (4*32 = 128)
  const int wn   = wave >> 2;  // 2 waves along N (2*64 = 128)
  const int m0   = blockIdx.y * 128;
  const int n0   = blockIdx.x * 128;

  const int laneHalf = lane >> 4;   // 0: lanes 0-15, 1: lanes 16-31
  const int lr       = lane & 15;

  int rowsA = M - m0; if (rowsA > 128) rowsA = 128;
  int rowsB = N - n0; if (rowsB > 128) rowsB = 128;
  const __bf16* Abase = A  + (size_t)m0 * K;
  const __bf16* Bbase = Bw + (size_t)n0 * K;

  const unsigned ldsA0 = (unsigned)(size_t)&lsA[0][0];
  const unsigned ldsA1 = (unsigned)(size_t)&lsA[1][0];
  const unsigned ldsB0 = (unsigned)(size_t)&lsB[0][0];
  const unsigned ldsB1 = (unsigned)(size_t)&lsB[1][0];

  v8f acc[2][4] = {};

  // Prologue: DMA first K tile into buffer 0 (one TDM issue per wavefront op)
  if (wave == 0) {
    tdm_load_tile(ldsA0, Abase, (unsigned)rowsA, (unsigned)K);
    tdm_load_tile(ldsB0, Bbase, (unsigned)rowsB, (unsigned)K);
  }

  int cur = 0;
  for (int kt = 0; kt < K; kt += 32) {
    // current buffer's DMA complete, then broadcast readiness (single barrier)
    if (wave == 0) __builtin_amdgcn_s_wait_tensorcnt(0);
    __syncthreads();
    // kick off next tile's DMA into the other buffer while we compute
    if (wave == 0 && (kt + 32) < K) {
      tdm_load_tile(cur ? ldsA0 : ldsA1, Abase + kt + 32, (unsigned)rowsA, (unsigned)K);
      tdm_load_tile(cur ? ldsB0 : ldsB1, Bbase + kt + 32, (unsigned)rowsB, (unsigned)K);
    }

    const __bf16* sa = &lsA[cur][0];
    const __bf16* sb = &lsB[cur][0];

    // ---- build fragments per CDNA5 ISA VGPR layouts -----------------------
    // A 16x32 bf16: VGPR j holds K = (j<4?0:16) + half*8 + 2*(j&3)
    v16bf af[2];
#pragma unroll
    for (int mt = 0; mt < 2; ++mt) {
      int row = wm * 32 + mt * 16 + lr;
      unsigned* dst = (unsigned*)&af[mt];
#pragma unroll
      for (int j = 0; j < 8; ++j) {
        int k = ((j < 4) ? 0 : 16) + laneHalf * 8 + (j & 3) * 2;
        dst[j] = *(const unsigned*)(&sa[row * LDS_STRIDE + k]);
      }
    }
    // B 32x16 bf16: VGPR j holds K = (j<4?0:16) + half*8 + 2*(j&3)
    v16bf bfr[4];
#pragma unroll
    for (int nt = 0; nt < 4; ++nt) {
      int col = wn * 64 + nt * 16 + lr;
      unsigned* dst = (unsigned*)&bfr[nt];
#pragma unroll
      for (int j = 0; j < 8; ++j) {
        int k = ((j < 4) ? 0 : 16) + laneHalf * 8 + (j & 3) * 2;
        dst[j] = *(const unsigned*)(&sb[col * LDS_STRIDE + k]);
      }
    }

    // ---- 8 WMMAs ----------------------------------------------------------
#pragma unroll
    for (int mt = 0; mt < 2; ++mt)
#pragma unroll
      for (int nt = 0; nt < 4; ++nt)
        acc[mt][nt] = __builtin_amdgcn_wmma_f32_16x16x32_bf16(
            false, af[mt], false, bfr[nt], (short)0, acc[mt][nt], false, false);

    cur ^= 1;
  }

  // ---- store C: VGPR r -> row (r + 8*half), lane -> column ----------------
#pragma unroll
  for (int mt = 0; mt < 2; ++mt) {
#pragma unroll
    for (int nt = 0; nt < 4; ++nt) {
      int rowBase = m0 + wm * 32 + mt * 16 + laneHalf * 8;
      int col     = n0 + wn * 64 + nt * 16 + lr;
      if (col < N) {
#pragma unroll
        for (int r = 0; r < 8; ++r) {
          int gr = rowBase + r;
          if (gr < M) C[(size_t)gr * N + col] = acc[mt][nt][r];
        }
      }
    }
  }
}

// ---------------------------------------------------------------------------
// Depthwise causal conv (K=4) + bias + SiLU. Emits fp32 + bf16 copies.
// xz layout: (BL, 4096); xc = cols [0,2048)
// ---------------------------------------------------------------------------
__global__ void conv_silu_kernel(const float* __restrict__ xz,
                                 const float* __restrict__ conv_w,
                                 const float* __restrict__ conv_b,
                                 float* __restrict__ xc_f,
                                 __bf16* __restrict__ xc_bf) {
  int idx = blockIdx.x * blockDim.x + threadIdx.x;
  if (idx >= BL_ * DI_) return;
  int d  = idx % DI_;
  int bl = idx / DI_;
  int l  = bl % L_;
  int b0 = bl - l;  // b*L
  float acc = conv_b[d];
#pragma unroll
  for (int j = 0; j < 4; ++j) {
    int ls = l - 3 + j;
    if (ls >= 0)
      acc += conv_w[d * 4 + j] * xz[(size_t)(b0 + ls) * (2 * DI_) + d];
  }
  float s = acc / (1.0f + __expf(-acc));
  xc_f[idx]  = s;
  xc_bf[idx] = (__bf16)s;
}

// ---------------------------------------------------------------------------
// Slice dBC[:, :64] -> bf16 for the delta projection GEMM
// ---------------------------------------------------------------------------
__global__ void slice_delta_bf16(const float* __restrict__ dBC,
                                 __bf16* __restrict__ dd) {
  int i = blockIdx.x * blockDim.x + threadIdx.x;
  if (i >= BL_ * DR_) return;
  int r = i >> 6, j = i & 63;
  dd[i] = (__bf16)dBC[r * (DR_ + 2 * DS_) + j];
}

// ---------------------------------------------------------------------------
// delta = softplus(delta_pre + b_dt), in place
// ---------------------------------------------------------------------------
__global__ void softplus_kernel(float* __restrict__ delta,
                                const float* __restrict__ b_dt) {
  int i = blockIdx.x * blockDim.x + threadIdx.x;
  if (i >= BL_ * DI_) return;
  float v = delta[i] + b_dt[i % DI_];
  delta[i] = (v > 20.0f) ? v : log1pf(__expf(v));
}

// ---------------------------------------------------------------------------
// Chunked selective scan (h_l = a_l*h_{l-1} + x_l is linear -> 3 phases).
// Chunk-state layout: buf[((b*NC + c)*DS + n)*DI + d]   (coalesced over d)
// ---------------------------------------------------------------------------
// Phase 1: per (b, chunk, d): decay product P[n] and zero-init local state S[n]
__global__ void scan_phase1(const float* __restrict__ delta,
                            const float* __restrict__ xc,
                            const float* __restrict__ dBC,
                            const float* __restrict__ A_log,
                            float* __restrict__ Pbuf,
                            float* __restrict__ Sbuf) {
  int idx = blockIdx.x * blockDim.x + threadIdx.x;
  if (idx >= B_ * NC_ * DI_) return;
  int d = idx % DI_;
  int t = idx / DI_;
  int c = t % NC_;
  int b = t / NC_;

  float Ad[DS_];
#pragma unroll
  for (int n = 0; n < DS_; ++n) Ad[n] = -__expf(A_log[d * DS_ + n]);

  float h[DS_], P[DS_];
#pragma unroll
  for (int n = 0; n < DS_; ++n) { h[n] = 0.0f; P[n] = 1.0f; }

  for (int l = c * CL_; l < (c + 1) * CL_; ++l) {
    size_t r  = (size_t)b * L_ + l;
    float  dt = delta[r * DI_ + d];
    float  xv = xc[r * DI_ + d];
    const float* row = dBC + r * (DR_ + 2 * DS_);
#pragma unroll
    for (int n = 0; n < DS_; ++n) {
      float a = __expf(dt * Ad[n]);
      h[n] = a * h[n] + dt * row[DR_ + n] * xv;
      P[n] *= a;
    }
  }
  size_t base = ((size_t)(b * NC_ + c) * DS_) * DI_ + d;
#pragma unroll
  for (int n = 0; n < DS_; ++n) {
    Pbuf[base + (size_t)n * DI_] = P[n];
    Sbuf[base + (size_t)n * DI_] = h[n];
  }
}

// Phase 2: per (b, d): stitch chunks; rewrite Pbuf with each chunk's H_init
__global__ void scan_phase2(float* __restrict__ Pbuf,
                            const float* __restrict__ Sbuf) {
  int idx = blockIdx.x * blockDim.x + threadIdx.x;
  if (idx >= B_ * DI_) return;
  int d = idx % DI_;
  int b = idx / DI_;

  float H[DS_];
#pragma unroll
  for (int n = 0; n < DS_; ++n) H[n] = 0.0f;

  for (int c = 0; c < NC_; ++c) {
    size_t base = ((size_t)(b * NC_ + c) * DS_) * DI_ + d;
#pragma unroll
    for (int n = 0; n < DS_; ++n) {
      size_t o = base + (size_t)n * DI_;
      float p = Pbuf[o];
      float s = Sbuf[o];
      Pbuf[o] = H[n];          // H_init for chunk c
      H[n] = p * H[n] + s;     // running end-state
    }
  }
}

// Phase 3: per (b, chunk, d): re-run chunk from H_init, emit gated output
__global__ void scan_phase3(const float* __restrict__ delta,
                            const float* __restrict__ xc,
                            const float* __restrict__ dBC,
                            const float* __restrict__ A_log,
                            const float* __restrict__ Dp,
                            const float* __restrict__ xz,
                            const float* __restrict__ Hinit,
                            __bf16* __restrict__ yz) {
  int idx = blockIdx.x * blockDim.x + threadIdx.x;
  if (idx >= B_ * NC_ * DI_) return;
  int d = idx % DI_;
  int t = idx / DI_;
  int c = t % NC_;
  int b = t / NC_;

  float Ad[DS_];
#pragma unroll
  for (int n = 0; n < DS_; ++n) Ad[n] = -__expf(A_log[d * DS_ + n]);
  float Dv = Dp[d];

  float h[DS_];
  size_t base = ((size_t)(b * NC_ + c) * DS_) * DI_ + d;
#pragma unroll
  for (int n = 0; n < DS_; ++n) h[n] = Hinit[base + (size_t)n * DI_];

  for (int l = c * CL_; l < (c + 1) * CL_; ++l) {
    size_t r  = (size_t)b * L_ + l;
    float  dt = delta[r * DI_ + d];
    float  xv = xc[r * DI_ + d];
    const float* row = dBC + r * (DR_ + 2 * DS_);
    float y = 0.0f;
#pragma unroll
    for (int n = 0; n < DS_; ++n) {
      float a = __expf(dt * Ad[n]);
      h[n] = a * h[n] + dt * row[DR_ + n] * xv;   // B part
      y   += h[n] * row[DR_ + DS_ + n];           // C part
    }
    y += Dv * xv;
    float z  = xz[r * (2 * DI_) + DI_ + d];
    float zs = z / (1.0f + __expf(-z));
    yz[r * DI_ + d] = (__bf16)(y * zs);
  }
}

// ---------------------------------------------------------------------------
// Host-side launcher
// ---------------------------------------------------------------------------
extern "C" void kernel_launch(void* const* d_in, const int* in_sizes, int n_in,
                              void* d_out, int out_size, void* d_ws, size_t ws_size,
                              hipStream_t stream) {
  (void)in_sizes; (void)n_in; (void)out_size; (void)ws_size;
  const float* x      = (const float*)d_in[0];
  const float* W_in   = (const float*)d_in[1];
  const float* conv_w = (const float*)d_in[2];
  const float* conv_b = (const float*)d_in[3];
  const float* W_x    = (const float*)d_in[4];
  const float* W_dt   = (const float*)d_in[5];
  const float* b_dt   = (const float*)d_in[6];
  const float* A_log  = (const float*)d_in[7];
  const float* Dp     = (const float*)d_in[8];
  const float* W_out  = (const float*)d_in[9];
  float* out = (float*)d_out;

  size_t off = 0;
  auto alloc = [&](size_t bytes) -> void* {
    off = (off + 255) & ~(size_t)255;
    void* p = (char*)d_ws + off;
    off += bytes;
    return p;
  };
  __bf16* xbf    = (__bf16*)alloc((size_t)BL_ * DM_ * 2);
  __bf16* Winbf  = (__bf16*)alloc((size_t)(2 * DI_) * DM_ * 2);
  __bf16* Wxbf   = (__bf16*)alloc((size_t)(DR_ + 2 * DS_) * DI_ * 2);
  __bf16* Wdtbf  = (__bf16*)alloc((size_t)DI_ * DR_ * 2);
  __bf16* Woutbf = (__bf16*)alloc((size_t)DM_ * DI_ * 2);
  float*  xz     = (float*)alloc((size_t)BL_ * (2 * DI_) * 4);
  float*  xcf    = (float*)alloc((size_t)BL_ * DI_ * 4);
  __bf16* xcbf   = (__bf16*)alloc((size_t)BL_ * DI_ * 2);
  float*  dBC    = (float*)alloc((size_t)BL_ * (DR_ + 2 * DS_) * 4);
  __bf16* ddbf   = (__bf16*)alloc((size_t)BL_ * DR_ * 2);
  float*  delta  = (float*)alloc((size_t)BL_ * DI_ * 4);
  __bf16* yzbf   = (__bf16*)alloc((size_t)BL_ * DI_ * 2);
  float*  Pbuf   = (float*)alloc((size_t)B_ * NC_ * DS_ * DI_ * 4);
  float*  Sbuf   = (float*)alloc((size_t)B_ * NC_ * DS_ * DI_ * 4);

  auto cvt = [&](const float* src, __bf16* dst, int n) {
    cvt_f32_to_bf16<<<(n + 255) / 256, 256, 0, stream>>>(src, dst, n);
  };
  cvt(x, xbf, BL_ * DM_);
  cvt(W_in, Winbf, 2 * DI_ * DM_);
  cvt(W_x, Wxbf, (DR_ + 2 * DS_) * DI_);
  cvt(W_dt, Wdtbf, DI_ * DR_);
  cvt(W_out, Woutbf, DM_ * DI_);

  auto gemm = [&](const __bf16* A, const __bf16* Bw, float* C, int M, int N, int K) {
    dim3 g((N + 127) / 128, (M + 127) / 128);
    wmma_gemm_bf16<<<g, 256, 0, stream>>>(A, Bw, C, M, N, K);
  };

  // 1) xz = x @ W_in.T         (2048 x 4096 x 1024)
  gemm(xbf, Winbf, xz, BL_, 2 * DI_, DM_);
  // 2) depthwise conv + SiLU
  conv_silu_kernel<<<(BL_ * DI_ + 255) / 256, 256, 0, stream>>>(xz, conv_w, conv_b, xcf, xcbf);
  // 3) dBC = xc @ W_x.T        (2048 x 96 x 2048)
  gemm(xcbf, Wxbf, dBC, BL_, DR_ + 2 * DS_, DI_);
  // 4) delta_pre = dBC[:, :64] @ W_dt.T   (2048 x 2048 x 64)
  slice_delta_bf16<<<(BL_ * DR_ + 255) / 256, 256, 0, stream>>>(dBC, ddbf);
  gemm(ddbf, Wdtbf, delta, BL_, DI_, DR_);
  // 5) delta = softplus(delta_pre + b_dt)
  softplus_kernel<<<(BL_ * DI_ + 255) / 256, 256, 0, stream>>>(delta, b_dt);
  // 6) chunked selective scan (3 phases) + Dp*xc + silu(z) gate -> bf16
  scan_phase1<<<(B_ * NC_ * DI_ + 255) / 256, 256, 0, stream>>>(delta, xcf, dBC, A_log, Pbuf, Sbuf);
  scan_phase2<<<(B_ * DI_ + 255) / 256, 256, 0, stream>>>(Pbuf, Sbuf);
  scan_phase3<<<(B_ * NC_ * DI_ + 255) / 256, 256, 0, stream>>>(delta, xcf, dBC, A_log, Dp, xz, Pbuf, yzbf);
  // 7) out = yz @ W_out.T      (2048 x 1024 x 2048)
  gemm(yzbf, Woutbf, out, BL_, DM_, DI_);
}
